// NativeLinear4bit_83253646066052
// MI455X (gfx1250) — compile-verified, matrix operand order checked by reading, not run on hardware
//
#include <hip/hip_runtime.h>
#include <hip/hip_bf16.h>

typedef __attribute__((ext_vector_type(16))) _Float16 v16h;
typedef __attribute__((ext_vector_type(8)))  _Float16 v8h;
typedef __attribute__((ext_vector_type(4)))  _Float16 v4h;
typedef __attribute__((ext_vector_type(8)))  float    v8f;

// Problem constants (from reference): IN_F = 4096, block_size = 64
#define INF      4096
#define HALF_IN  2048      // packed int32 words per output row (1 byte value each)
#define NBLK     64        // absmax blocks per row
#define KTILES   (INF / 32)

#define BM 128
#define BN 128
#define LDP 40             // LDS row pitch in halves (80 bytes: 16B-aligned, bank-conflict-free)

__global__ __launch_bounds__(256)
void nf4_wmma_gemm(const float* __restrict__ x,
                   const int*   __restrict__ packed,
                   const float* __restrict__ absmax,
                   const float* __restrict__ qmap,
                   const float* __restrict__ bias,
                   float* __restrict__ out,
                   int T, int OUTF) {
    __shared__ _Float16 Xs[BM * LDP];      // 128 x 32 f16 tile of X
    __shared__ _Float16 Ws[BN * LDP];      // 128 x 32 f16 dequantized W tile
    __shared__ float2   lut2[256];         // byte -> (nf4[hi], nf4[lo])

    const int tid    = threadIdx.x;
    const int oBase  = blockIdx.x * BN;    // output-feature tile base (N)
    const int mBase  = blockIdx.y * BM;    // x-row tile base (M)
    const int wave   = tid >> 5;
    const int lane   = tid & 31;
    const int laneHi = (lane >> 4) & 1;
    const int lr     = lane & 15;
    const int mWave  = (wave & 1) * 64;    // 2 waves along M, 64 rows each
    const int nWave  = (wave >> 1) * 32;   // 4 waves along N, 32 cols each

    // Build full-byte dequant table once (high nibble first, matching reference)
    {
        float hi = qmap[(tid >> 4) & 15];
        float lo = qmap[tid & 15];
        lut2[tid] = make_float2(hi, lo);
    }

    v8f acc[4][2] = {};

    for (int kt = 0; kt < KTILES; ++kt) {
        // ---- global loads into registers ----
        float4 xv[4];
        int4   wv[2];
        float  sc[2];
#pragma unroll
        for (int p = 0; p < 4; ++p) {
            int it = tid + p * 256;
            int xr = it >> 3, xc = it & 7;                 // 128 rows x 8 float4 chunks
            xv[p] = *(const float4*)(x + (size_t)(mBase + xr) * INF + kt * 32 + xc * 4);
        }
#pragma unroll
        for (int p = 0; p < 2; ++p) {
            int it = tid + p * 256;
            int wr = it >> 2, wc = it & 3;                 // 128 rows x 4 int4 chunks
            wv[p] = *(const int4*)(packed + (size_t)(oBase + wr) * HALF_IN + kt * 16 + wc * 4);
            sc[p] = absmax[(size_t)(oBase + wr) * NBLK + (kt >> 1)];
        }
        if (kt + 1 < KTILES) {
            // hint next tiles into cache (global_prefetch_b8)
            int xr = tid >> 3, xc = tid & 7;
            __builtin_prefetch(x + (size_t)(mBase + xr) * INF + (kt + 1) * 32 + xc * 4, 0, 0);
            int wr = tid >> 2, wc = tid & 3;
            __builtin_prefetch(packed + (size_t)(oBase + wr) * HALF_IN + (kt + 1) * 16 + wc * 4, 0, 0);
        }

        __syncthreads();   // previous iteration's LDS reads done (also fences lut2 on iter 0)

        // ---- stage into LDS (convert / dequantize) ----
#pragma unroll
        for (int p = 0; p < 4; ++p) {
            int it = tid + p * 256;
            int xr = it >> 3, xc = it & 7;
            v4h hx;
            hx[0] = (_Float16)xv[p].x;
            hx[1] = (_Float16)xv[p].y;
            hx[2] = (_Float16)xv[p].z;
            hx[3] = (_Float16)xv[p].w;
            *(v4h*)&Xs[xr * LDP + xc * 4] = hx;
        }
#pragma unroll
        for (int p = 0; p < 2; ++p) {
            int it = tid + p * 256;
            int wr = it >> 2, wc = it & 3;
            const int* pb = (const int*)&wv[p];
            v8h hw;
#pragma unroll
            for (int q = 0; q < 4; ++q) {
                float2 lv = lut2[pb[q] & 0xFF];            // one ds_load_b64 per byte
                hw[2 * q]     = (_Float16)(lv.x * sc[p]);
                hw[2 * q + 1] = (_Float16)(lv.y * sc[p]);
            }
            *(v8h*)&Ws[wr * LDP + wc * 8] = hw;
        }

        __syncthreads();

        // ---- WMMA fragment loads (ISA 16-bit A-layout) ----
        const int kOff = laneHi ? 8 : 0;
        v16h af[4], bf[2];
#pragma unroll
        for (int mi = 0; mi < 4; ++mi) {
            const _Float16* pA = &Xs[(mWave + mi * 16 + lr) * LDP + kOff];
            v8h lo = *(const v8h*)pA;
            v8h hi = *(const v8h*)(pA + 16);
            af[mi] = __builtin_shufflevector(lo, hi, 0,1,2,3,4,5,6,7,8,9,10,11,12,13,14,15);
        }
#pragma unroll
        for (int ni = 0; ni < 2; ++ni) {
            const _Float16* pB = &Ws[(nWave + ni * 16 + lr) * LDP + kOff];
            v8h lo = *(const v8h*)pB;
            v8h hi = *(const v8h*)(pB + 16);
            bf[ni] = __builtin_shufflevector(lo, hi, 0,1,2,3,4,5,6,7,8,9,10,11,12,13,14,15);
        }

#pragma unroll
        for (int mi = 0; mi < 4; ++mi)
#pragma unroll
            for (int ni = 0; ni < 2; ++ni)
                acc[mi][ni] = __builtin_amdgcn_wmma_f32_16x16x32_f16(
                    false, af[mi], false, bf[ni], (short)0, acc[mi][ni], false, false);
    }

    // ---- epilogue: D layout => VGPR r: lanes 0-15: (M=r, N=lane), lanes 16-31: (M=8+r, N=lane-16)
#pragma unroll
    for (int ni = 0; ni < 2; ++ni) {
        int n = oBase + nWave + ni * 16 + lr;
        float bv = bias[n];
#pragma unroll
        for (int mi = 0; mi < 4; ++mi) {
            int m0 = mBase + mWave + mi * 16 + laneHi * 8;
#pragma unroll
            for (int r = 0; r < 8; ++r)
                out[(size_t)(m0 + r) * OUTF + n] = acc[mi][ni][r] + bv;
        }
    }
}

extern "C" void kernel_launch(void* const* d_in, const int* in_sizes, int n_in,
                              void* d_out, int out_size, void* d_ws, size_t ws_size,
                              hipStream_t stream) {
    const float* x      = (const float*)d_in[0];
    const int*   packed = (const int*)d_in[1];
    const float* absmax = (const float*)d_in[2];
    const float* qmap   = (const float*)d_in[3];
    const float* bias   = (const float*)d_in[4];
    float*       out    = (float*)d_out;

    const int T    = in_sizes[0] / INF;   // 2048
    const int OUTF = in_sizes[4];         // 11008

    dim3 grid((OUTF + BN - 1) / BN, (T + BM - 1) / BM);  // 86 x 16
    dim3 block(256);
    nf4_wmma_gemm<<<grid, block, 0, stream>>>(x, packed, absmax, qmap, bias, out, T, OUTF);
}
